// MultiHeadAttention_55035710931465
// MI455X (gfx1250) — matrix operand order repaired and checked by reading, not compiled, not run on hardware
//
#include <hip/hip_runtime.h>

// ---------- constants ----------
#define B_  2
#define L_  2048
#define DM_ 1024
#define H_  16
#define DK_ 64
#define M4_ 4096   // B_*L_

typedef __attribute__((ext_vector_type(16))) __bf16 v16bf;
typedef __attribute__((ext_vector_type(8)))  __bf16 v8bf;
typedef __attribute__((ext_vector_type(8)))  float  v8f;

// ---------- helpers ----------
static __device__ __forceinline__ __bf16 cvt_bf16(float f) {
    unsigned u = __builtin_bit_cast(unsigned, f);
    u += 0x7fffu + ((u >> 16) & 1u);               // round-to-nearest-even
    unsigned short h = (unsigned short)(u >> 16);
    return __builtin_bit_cast(__bf16, h);
}

static __device__ __forceinline__ v8f wmma_bf16(v16bf a, v16bf b, v8f c) {
    // D = A(16x32 bf16) * B(32x16 bf16) + C(16x16 f32)
    return __builtin_amdgcn_wmma_f32_16x16x32_bf16(
        false, a, false, b, (short)0, c, false, false);
}

// A fragment, 16x32 bf16, row-major src, row stride lda (elements).
static __device__ __forceinline__ v16bf load_a_bf16(const __bf16* src, int lda,
                                                    int m0, int k0, int lane) {
    int row = m0 + (lane & 15);
    int kb  = k0 + ((lane >> 4) << 3);
    const __bf16* p = src + (size_t)row * lda + kb;
    v8bf lo = *reinterpret_cast<const v8bf*>(p);
    v8bf hi = *reinterpret_cast<const v8bf*>(p + 16);
    v16bf r;
#pragma unroll
    for (int i = 0; i < 8; ++i) { r[i] = lo[i]; r[i + 8] = hi[i]; }
    return r;
}

// A fragment from f32 source, converted to bf16 in registers (for attn @ V).
static __device__ __forceinline__ v16bf load_a_f32(const float* src, int lda,
                                                   int m0, int k0, int lane) {
    int row = m0 + (lane & 15);
    int kb  = k0 + ((lane >> 4) << 3);
    const float* p = src + (size_t)row * lda + kb;
    v8f lo = *reinterpret_cast<const v8f*>(p);
    v8f hi = *reinterpret_cast<const v8f*>(p + 16);
    v16bf r;
#pragma unroll
    for (int i = 0; i < 8; ++i) { r[i] = cvt_bf16(lo[i]); r[i + 8] = cvt_bf16(hi[i]); }
    return r;
}

// B fragment, 32x16 bf16 from B-transposed storage BT[N][K], row stride ldb.
static __device__ __forceinline__ v16bf load_b_bt(const __bf16* bt, int ldb,
                                                  int n0, int k0, int lane) {
    int n  = n0 + (lane & 15);
    int kb = k0 + ((lane >> 4) << 3);
    const __bf16* p = bt + (size_t)n * ldb + kb;
    v8bf lo = *reinterpret_cast<const v8bf*>(p);
    v8bf hi = *reinterpret_cast<const v8bf*>(p + 16);
    v16bf r;
#pragma unroll
    for (int i = 0; i < 8; ++i) { r[i] = lo[i]; r[i + 8] = hi[i]; }
    return r;
}

// C tile (16x16 f32) store helpers. lane = N%16, VGPR r -> M = r + 8*(lane/16).
static __device__ __forceinline__ void store_c_f32(float* dst, int ldc, int m0, int n0,
                                                   int lane, v8f c, float scale, float bias) {
    int n  = n0 + (lane & 15);
    int mb = m0 + ((lane >> 4) << 3);
#pragma unroll
    for (int r = 0; r < 8; ++r)
        dst[(size_t)(mb + r) * ldc + n] = c[r] * scale + bias;
}

static __device__ __forceinline__ void store_c_bf16(__bf16* dst, int ldc, int m0, int n0,
                                                    int lane, v8f c) {
    int n  = n0 + (lane & 15);
    int mb = m0 + ((lane >> 4) << 3);
#pragma unroll
    for (int r = 0; r < 8; ++r)
        dst[(size_t)(mb + r) * ldc + n] = cvt_bf16(c[r]);
}

// ---------- elementwise casts ----------
__global__ void k_cast_bf16(const float* __restrict__ in, __bf16* __restrict__ out, int n) {
    int i = blockIdx.x * blockDim.x + threadIdx.x;
    int s = gridDim.x * blockDim.x;
    for (; i < n; i += s) out[i] = cvt_bf16(in[i]);
}

// in: [batch][rows][cols] f32  ->  out: [batch][cols][rows] bf16
__global__ void k_transpose_cast(const float* __restrict__ in, __bf16* __restrict__ out,
                                 int rows, int cols, int n) {
    int i = blockIdx.x * blockDim.x + threadIdx.x;
    int s = gridDim.x * blockDim.x;
    int rc = rows * cols;
    for (; i < n; i += s) {
        int b = i / rc, rem = i - b * rc;
        int r = rem / cols, c = rem - r * cols;
        out[(size_t)b * rc + (size_t)c * rows + r] = cvt_bf16(in[i]);
    }
}

// ---------- QKV projection: per head, [4096x1024] x [1024x64] ----------
// grid(32, H_, 3), block 256 (8 waves); wave -> 16x64 tile.
// 2x-unrolled double-buffered K loop: alternating register sets, no copies.
__global__ void k_proj_qkv(const __bf16* __restrict__ qbf, const __bf16* __restrict__ kbf,
                           const __bf16* __restrict__ vbf,
                           const __bf16* __restrict__ wqt, const __bf16* __restrict__ wkt,
                           const __bf16* __restrict__ wvt,
                           __bf16* __restrict__ qs, __bf16* __restrict__ ks,
                           __bf16* __restrict__ vt) {
    int h = blockIdx.y, which = blockIdx.z;
    int wave = threadIdx.x >> 5, lane = threadIdx.x & 31;
    int m0 = (blockIdx.x * 8 + wave) * 16;

    const __bf16* A  = (which == 0) ? qbf : (which == 1) ? kbf : vbf;
    const __bf16* BT = ((which == 0) ? wqt : (which == 1) ? wkt : wvt) + (size_t)h * DK_ * DM_;

    v8f acc[4] = {};
    v16bf a0 = load_a_bf16(A, DM_, m0, 0, lane);
    v16bf b0[4], a1, b1[4];
#pragma unroll
    for (int nt = 0; nt < 4; ++nt) b0[nt] = load_b_bt(BT, DM_, nt * 16, 0, lane);

    for (int k0 = 0; k0 < DM_; k0 += 64) {
        // prefetch odd step (k0+32 always < DM_)
        a1 = load_a_bf16(A, DM_, m0, k0 + 32, lane);
#pragma unroll
        for (int nt = 0; nt < 4; ++nt) b1[nt] = load_b_bt(BT, DM_, nt * 16, k0 + 32, lane);
#pragma unroll
        for (int nt = 0; nt < 4; ++nt) acc[nt] = wmma_bf16(a0, b0[nt], acc[nt]);
        // prefetch next even step
        if (k0 + 64 < DM_) {
            a0 = load_a_bf16(A, DM_, m0, k0 + 64, lane);
#pragma unroll
            for (int nt = 0; nt < 4; ++nt) b0[nt] = load_b_bt(BT, DM_, nt * 16, k0 + 64, lane);
        }
#pragma unroll
        for (int nt = 0; nt < 4; ++nt) acc[nt] = wmma_bf16(a1, b1[nt], acc[nt]);
    }

    if (which < 2) {
        __bf16* dst = ((which == 0) ? qs : ks) + (size_t)h * M4_ * DK_;
#pragma unroll
        for (int nt = 0; nt < 4; ++nt)
            store_c_bf16(dst, DK_, m0, nt * 16, lane, acc[nt]);
    } else {
        __bf16* dst = vt + (size_t)h * B_ * DK_ * L_;
#pragma unroll
        for (int nt = 0; nt < 4; ++nt) {
            int n  = nt * 16 + (lane & 15);
            int mb = m0 + ((lane >> 4) << 3);
#pragma unroll
            for (int r = 0; r < 8; ++r) {
                int m = mb + r, b = m >> 11, l = m & (L_ - 1);
                dst[(size_t)b * DK_ * L_ + (size_t)n * L_ + l] = cvt_bf16(acc[nt][r]);
            }
        }
    }
}

// ---------- fused scores + softmax ----------
// One block = 16 rows x 2048 cols of S for one (h,b); 8 waves, each wave owns a
// 16x256 strip (16 accum tiles, 128 VGPRs/lane). Row max/sum: fold 16 cols in-lane,
// half-wave shuffle reduce, cross-wave via LDS. Writes normalized softmax once.
// grid(L_/16, 32), block 256.
__global__ void k_scores_softmax(const __bf16* __restrict__ qs, const __bf16* __restrict__ ks,
                                 float* __restrict__ attn) {
    __shared__ float red[8][16];
    __shared__ float rowred[16];

    int z = blockIdx.y, h = z >> 1, b = z & 1;
    int wave = threadIdx.x >> 5, lane = threadIdx.x & 31;
    int half = lane >> 4;
    int m0 = blockIdx.x * 16;
    int n0w = wave * 256;

    const __bf16* Q = qs + ((size_t)h * M4_ + (size_t)b * L_) * DK_;
    const __bf16* K = ks + ((size_t)h * M4_ + (size_t)b * L_) * DK_;
    float* S = attn + (size_t)z * L_ * L_;

    v8f acc[16] = {};
#pragma unroll
    for (int k0 = 0; k0 < DK_; k0 += 32) {
        v16bf a = load_a_bf16(Q, DK_, m0, k0, lane);
#pragma unroll
        for (int nt = 0; nt < 16; ++nt) {
            v16bf bb = load_b_bt(K, DK_, n0w + nt * 16, k0, lane);
            acc[nt] = wmma_bf16(a, bb, acc[nt]);
        }
    }

    // scale by 1/sqrt(d_model) and fold per-lane row maxima over the 16 columns
    const float scale = 0.03125f;
    float rm[8];
#pragma unroll
    for (int r = 0; r < 8; ++r) rm[r] = -3.4e38f;
#pragma unroll
    for (int nt = 0; nt < 16; ++nt)
#pragma unroll
        for (int r = 0; r < 8; ++r) {
            float v = acc[nt][r] * scale;
            acc[nt][r] = v;
            rm[r] = fmaxf(rm[r], v);
        }
#pragma unroll
    for (int o = 1; o < 16; o <<= 1)
#pragma unroll
        for (int r = 0; r < 8; ++r) rm[r] = fmaxf(rm[r], __shfl_xor(rm[r], o, 32));
    if ((lane & 15) == 0) {
#pragma unroll
        for (int r = 0; r < 8; ++r) red[wave][half * 8 + r] = rm[r];
    }
    __syncthreads();
    if (threadIdx.x < 16) {
        float m = red[0][threadIdx.x];
#pragma unroll
        for (int j = 1; j < 8; ++j) m = fmaxf(m, red[j][threadIdx.x]);
        rowred[threadIdx.x] = m;
    }
    __syncthreads();
    float rmax[8];
#pragma unroll
    for (int r = 0; r < 8; ++r) rmax[r] = rowred[half * 8 + r];

    // exp + per-lane row sums
    float rs[8];
#pragma unroll
    for (int r = 0; r < 8; ++r) rs[r] = 0.0f;
#pragma unroll
    for (int nt = 0; nt < 16; ++nt)
#pragma unroll
        for (int r = 0; r < 8; ++r) {
            float e = __expf(acc[nt][r] - rmax[r]);
            acc[nt][r] = e;
            rs[r] += e;
        }
#pragma unroll
    for (int o = 1; o < 16; o <<= 1)
#pragma unroll
        for (int r = 0; r < 8; ++r) rs[r] += __shfl_xor(rs[r], o, 32);
    __syncthreads();  // everyone done reading rowred(max) before reuse
    if ((lane & 15) == 0) {
#pragma unroll
        for (int r = 0; r < 8; ++r) red[wave][half * 8 + r] = rs[r];
    }
    __syncthreads();
    if (threadIdx.x < 16) {
        float s = red[0][threadIdx.x];
#pragma unroll
        for (int j = 1; j < 8; ++j) s += red[j][threadIdx.x];
        rowred[threadIdx.x] = 1.0f / s;
    }
    __syncthreads();
    float rinv[8];
#pragma unroll
    for (int r = 0; r < 8; ++r) rinv[r] = rowred[half * 8 + r];

    // store normalized attention (single write of the 537 MB tensor)
    int n  = lane & 15;
    int mb = m0 + half * 8;
#pragma unroll
    for (int nt = 0; nt < 16; ++nt) {
        int col = n0w + nt * 16 + n;
#pragma unroll
        for (int r = 0; r < 8; ++r)
            S[(size_t)(mb + r) * L_ + col] = acc[nt][r] * rinv[r];
    }
}

// ---------- attn @ V -> concat (bf16, head-major feature concat) ----------
// grid(16, 32): x -> 128-row block, y -> h*2+b; 2x-unrolled double-buffered K loop.
__global__ void k_attnv(const float* __restrict__ attn, const __bf16* __restrict__ vt,
                        __bf16* __restrict__ concat) {
    int z = blockIdx.y, h = z >> 1, b = z & 1;
    int wave = threadIdx.x >> 5, lane = threadIdx.x & 31;
    int m0 = (blockIdx.x * 8 + wave) * 16;

    const float*  Ap = attn + (size_t)z * L_ * L_;
    const __bf16* Vt = vt + (size_t)z * DK_ * L_;

    v8f acc[4] = {};
    v16bf a0 = load_a_f32(Ap, L_, m0, 0, lane);
    v16bf b0[4], a1, b1[4];
#pragma unroll
    for (int nt = 0; nt < 4; ++nt) b0[nt] = load_b_bt(Vt, L_, nt * 16, 0, lane);

    for (int k0 = 0; k0 < L_; k0 += 64) {
        a1 = load_a_f32(Ap, L_, m0, k0 + 32, lane);
#pragma unroll
        for (int nt = 0; nt < 4; ++nt) b1[nt] = load_b_bt(Vt, L_, nt * 16, k0 + 32, lane);
#pragma unroll
        for (int nt = 0; nt < 4; ++nt) acc[nt] = wmma_bf16(a0, b0[nt], acc[nt]);
        if (k0 + 64 < L_) {
            a0 = load_a_f32(Ap, L_, m0, k0 + 64, lane);
#pragma unroll
            for (int nt = 0; nt < 4; ++nt) b0[nt] = load_b_bt(Vt, L_, nt * 16, k0 + 64, lane);
        }
#pragma unroll
        for (int nt = 0; nt < 4; ++nt) acc[nt] = wmma_bf16(a1, b1[nt], acc[nt]);
    }

    // store to concat[b*L + l][h*64 + n]
#pragma unroll
    for (int nt = 0; nt < 4; ++nt) {
        int n  = h * DK_ + nt * 16 + (lane & 15);
        int mb = m0 + ((lane >> 4) << 3);
#pragma unroll
        for (int r = 0; r < 8; ++r)
            concat[((size_t)b * L_ + (mb + r)) * DM_ + n] = cvt_bf16(acc[nt][r]);
    }
}

// ---------- output projection: [4096x1024] x [1024x1024] + bias ----------
// grid(16, 32): x -> 64-col block, y -> 128-row block; 2x-unrolled double buffer.
__global__ void k_outproj(const __bf16* __restrict__ concat, const __bf16* __restrict__ pwt,
                          const float* __restrict__ bias, float* __restrict__ projout) {
    int wave = threadIdx.x >> 5, lane = threadIdx.x & 31;
    int m0 = (blockIdx.y * 8 + wave) * 16;
    int n0 = blockIdx.x * 64;

    v8f acc[4] = {};
    v16bf a0 = load_a_bf16(concat, DM_, m0, 0, lane);
    v16bf b0[4], a1, b1[4];
#pragma unroll
    for (int nt = 0; nt < 4; ++nt) b0[nt] = load_b_bt(pwt, DM_, n0 + nt * 16, 0, lane);

    for (int k0 = 0; k0 < DM_; k0 += 64) {
        a1 = load_a_bf16(concat, DM_, m0, k0 + 32, lane);
#pragma unroll
        for (int nt = 0; nt < 4; ++nt) b1[nt] = load_b_bt(pwt, DM_, n0 + nt * 16, k0 + 32, lane);
#pragma unroll
        for (int nt = 0; nt < 4; ++nt) acc[nt] = wmma_bf16(a0, b0[nt], acc[nt]);
        if (k0 + 64 < DM_) {
            a0 = load_a_bf16(concat, DM_, m0, k0 + 64, lane);
#pragma unroll
            for (int nt = 0; nt < 4; ++nt) b0[nt] = load_b_bt(pwt, DM_, n0 + nt * 16, k0 + 64, lane);
        }
#pragma unroll
        for (int nt = 0; nt < 4; ++nt) acc[nt] = wmma_bf16(a1, b1[nt], acc[nt]);
    }

#pragma unroll
    for (int nt = 0; nt < 4; ++nt) {
        float bv = bias[n0 + nt * 16 + (lane & 15)];
        store_c_f32(projout, DM_, m0, n0 + nt * 16, lane, acc[nt], 1.0f, bv);
    }
}

// ---------- LayerNorm (unbiased std, eps on sigma) ----------
__global__ void k_layernorm(const float* __restrict__ x, const float* __restrict__ ln_a,
                            const float* __restrict__ ln_b, float* __restrict__ out) {
    __shared__ float sm1[8], sm2[8];
    __shared__ float bc1, bc2;
    const float* row = x + (size_t)blockIdx.x * DM_;
    float* orow = out + (size_t)blockIdx.x * DM_;
    int t = threadIdx.x, w = t >> 5;

    float v[4], s = 0.0f, s2 = 0.0f;
#pragma unroll
    for (int i = 0; i < 4; ++i) {
        v[i] = row[t + i * 256];
        s += v[i]; s2 += v[i] * v[i];
    }
#pragma unroll
    for (int o = 16; o > 0; o >>= 1) { s += __shfl_xor(s, o, 32); s2 += __shfl_xor(s2, o, 32); }
    if ((t & 31) == 0) { sm1[w] = s; sm2[w] = s2; }
    __syncthreads();
    if (t == 0) {
        float a = 0.0f, b2 = 0.0f;
#pragma unroll
        for (int j = 0; j < 8; ++j) { a += sm1[j]; b2 += sm2[j]; }
        bc1 = a; bc2 = b2;
    }
    __syncthreads();
    float mu  = bc1 * (1.0f / DM_);
    float var = (bc2 - (float)DM_ * mu * mu) * (1.0f / (DM_ - 1));
    float sigma = sqrtf(fmaxf(var, 0.0f));
    float inv = 1.0f / (sigma + 1e-3f);
#pragma unroll
    for (int i = 0; i < 4; ++i) {
        int c = t + i * 256;
        orow[c] = (v[i] - mu) * inv * ln_a[c] + ln_b[c];
    }
}

// ---------- launcher ----------
extern "C" void kernel_launch(void* const* d_in, const int* in_sizes, int n_in,
                              void* d_out, int out_size, void* d_ws, size_t ws_size,
                              hipStream_t stream) {
    const float* q      = (const float*)d_in[0];
    const float* k      = (const float*)d_in[1];
    const float* v      = (const float*)d_in[2];
    const float* w_qs   = (const float*)d_in[3];
    const float* w_ks   = (const float*)d_in[4];
    const float* w_vs   = (const float*)d_in[5];
    const float* proj_w = (const float*)d_in[6];
    const float* proj_b = (const float*)d_in[7];
    const float* ln_a   = (const float*)d_in[8];
    const float* ln_b   = (const float*)d_in[9];

    float* out  = (float*)d_out;                       // [2,2048,1024]
    float* attn = out + (size_t)M4_ * DM_;             // [32,2048,2048]

    // workspace carve (bytes)
    char* w = (char*)d_ws;
    size_t off = 0;
    auto carve = [&](size_t bytes) { char* p = w + off; off += (bytes + 255) & ~(size_t)255; return p; };
    __bf16* qbf     = (__bf16*)carve((size_t)M4_ * DM_ * 2);
    __bf16* kbf     = (__bf16*)carve((size_t)M4_ * DM_ * 2);
    __bf16* vbf     = (__bf16*)carve((size_t)M4_ * DM_ * 2);
    __bf16* wqt     = (__bf16*)carve((size_t)H_ * DK_ * DM_ * 2);     // [h][64][1024]
    __bf16* wkt     = (__bf16*)carve((size_t)H_ * DK_ * DM_ * 2);
    __bf16* wvt     = (__bf16*)carve((size_t)H_ * DK_ * DM_ * 2);
    __bf16* pwt     = (__bf16*)carve((size_t)DM_ * DM_ * 2);          // [out][in]
    __bf16* qs      = (__bf16*)carve((size_t)H_ * M4_ * DK_ * 2);     // [h][4096][64]
    __bf16* ks      = (__bf16*)carve((size_t)H_ * M4_ * DK_ * 2);
    __bf16* vt      = (__bf16*)carve((size_t)H_ * B_ * DK_ * L_ * 2); // [h][b][64][2048]
    __bf16* concat  = (__bf16*)carve((size_t)M4_ * DM_ * 2);          // [4096][1024]
    float*  projout = (float*)carve((size_t)M4_ * DM_ * 4);           // [4096][1024]

    const int nqkv = M4_ * DM_;
    const int nw   = H_ * DM_ * DK_;

    // 1) casts / B-transposes
    k_cast_bf16<<<2048, 256, 0, stream>>>(q, qbf, nqkv);
    k_cast_bf16<<<2048, 256, 0, stream>>>(k, kbf, nqkv);
    k_cast_bf16<<<2048, 256, 0, stream>>>(v, vbf, nqkv);
    k_transpose_cast<<<2048, 256, 0, stream>>>(w_qs, wqt, DM_, DK_, nw);
    k_transpose_cast<<<2048, 256, 0, stream>>>(w_ks, wkt, DM_, DK_, nw);
    k_transpose_cast<<<2048, 256, 0, stream>>>(w_vs, wvt, DM_, DK_, nw);
    k_transpose_cast<<<2048, 256, 0, stream>>>(proj_w, pwt, DM_, DM_, DM_ * DM_);

    // 2) per-head QKV projections (WMMA bf16, ping-pong double buffer)
    k_proj_qkv<<<dim3(32, H_, 3), 256, 0, stream>>>(qbf, kbf, vbf, wqt, wkt, wvt, qs, ks, vt);

    // 3) fused scores + softmax -> attns region (single write of attention tensor)
    k_scores_softmax<<<dim3(L_ / 16, 32), 256, 0, stream>>>(qs, ks, attn);

    // 4) attn @ V -> head-concat bf16 (ping-pong double buffer)
    k_attnv<<<dim3(16, 32), 256, 0, stream>>>(attn, vt, concat);

    // 5) output projection + bias (ping-pong double buffer)
    k_outproj<<<dim3(16, 32), 256, 0, stream>>>(concat, pwt, proj_b, projout);

    // 6) LayerNorm -> out region
    k_layernorm<<<M4_, 256, 0, stream>>>(projout, ln_a, ln_b, out);
}
